// Unified2DAttentionOnFrame_85950885528317
// MI455X (gfx1250) — matrix operand order repaired and checked by reading, not compiled, not run on hardware
//
#include <hip/hip_runtime.h>
#include <hip/hip_bf16.h>

// MI455X (gfx1250) implementation.
// Roofline: ~52 GFLOP total, dominated by attention GEMMs (34 GF); all live
// tensors fit in the 192MB L2 (HBM floor ~134MB @ 23.3TB/s ~ 6us) ->
// compute-bound. Path: f16 WMMA (v_wmma_f32_16x16x32_f16, f32 accumulate) for
// every GEMM. B operands staged TRANSPOSED in LDS with padded row stride so
// each fragment is two ds_load_b128 per lane (no scalar gather / v_mov_b16
// repacking, no systematic bank conflicts). Q/K tiles of the attention kernel
// are fetched by the Tensor Data Mover (tensor_load_to_lds + s_wait_tensorcnt).

typedef __attribute__((ext_vector_type(16))) _Float16 v16h;
typedef __attribute__((ext_vector_type(8)))  _Float16 v8h;
typedef __attribute__((ext_vector_type(8)))  float    v8f;
typedef __attribute__((ext_vector_type(4)))  unsigned int v4u;
typedef __attribute__((ext_vector_type(8)))  int      v8i;
typedef __attribute__((ext_vector_type(4)))  int      v4i;

#define B_    4
#define CIN_  128
#define FREQ_ 64
#define T_    512
#define H_    8
#define OC3_  384
#define HD_   1024   // per-head feature dim = 16 channels * 64 freq

#define LDW_  136    // padded row stride (halves) for 128-wide tiles: 272B, 16B-aligned
#define LDV_  72     // padded row stride (halves) for 64-wide tiles: 144B, 16B-aligned

// ---------------- WMMA fragment helpers (wave32 layouts per CDNA5 ISA 7.12.2)

// A matrix 16x32 f16 from row-major [M][ld]: lane L holds row M=L&15,
// K base (L>=16)*8; halves {K..K+7, K+16..K+23} -> two 16B LDS loads.
__device__ __forceinline__ v16h frag_a(const _Float16* base, int ld) {
  int lane = threadIdx.x & 31;
  const v8h* p = (const v8h*)(base + (lane & 15) * ld + ((lane >> 4) << 3));
  v8h lo = p[0];
  v8h hi = p[2];   // +16 halves = +32B
  return __builtin_shufflevector(lo, hi, 0, 1, 2, 3, 4, 5, 6, 7,
                                 8, 9, 10, 11, 12, 13, 14, 15);
}

// B matrix 32x16 f16 from TRANSPOSED storage [N][ld] (B[k][n] = base[n*ld+k]):
// lane L reads row N=L&15, K base (L>=16)*8; halves {K..K+7, K+16..K+23}.
__device__ __forceinline__ v16h frag_bT(const _Float16* base, int ld) {
  int lane = threadIdx.x & 31;
  const v8h* p = (const v8h*)(base + (lane & 15) * ld + ((lane >> 4) << 3));
  v8h lo = p[0];
  v8h hi = p[2];
  return __builtin_shufflevector(lo, hi, 0, 1, 2, 3, 4, 5, 6, 7,
                                 8, 9, 10, 11, 12, 13, 14, 15);
}

__device__ __forceinline__ v8f wmma_f16(v16h a, v16h b, v8f c) {
  return __builtin_amdgcn_wmma_f32_16x16x32_f16(false, a, false, b,
                                                (short)0, c, false, false);
}

__device__ __forceinline__ float wave_sum(float v) {
#pragma unroll
  for (int o = 16; o > 0; o >>= 1) v += __shfl_xor(v, o, 32);
  return v;
}
__device__ __forceinline__ float wave_max(float v) {
#pragma unroll
  for (int o = 16; o > 0; o >>= 1) v = fmaxf(v, __shfl_xor(v, o, 32));
  return v;
}

// ---------------- Tensor Data Mover: 2D f16 tile -> LDS (CDNA5 ISA ch.8 D#)
#if defined(__has_builtin) && __has_builtin(__builtin_amdgcn_tensor_load_to_lds)
#define USE_TDM 1
#else
#define USE_TDM 0
#endif

#if USE_TDM
__device__ __forceinline__ void tdm_load_2d_f16(unsigned lds_off,
                                                const void* gaddr,
                                                unsigned tile_w, unsigned tile_h,
                                                unsigned stride_elems) {
  unsigned long long ga = (unsigned long long)(uintptr_t)gaddr;
  v4u g0;
  g0[0] = 1u;                                   // count=1, user descriptor
  g0[1] = lds_off;                              // lds_addr (bytes)
  g0[2] = (unsigned)(ga & 0xffffffffu);         // global_addr[31:0]
  g0[3] = (unsigned)((ga >> 32) & 0x1ffffffu)   // global_addr[56:32]
          | (2u << 30);                         // type=2 (image)
  const unsigned td0 = 1u << 20, td1 = 1u << 20;  // huge tensor dims: no clip
  v8i g1;
  g1[0] = (int)(1u << 16);                      // data_size=2B, mask=0, no pad
  g1[1] = (int)((td0 & 0xffffu) << 16);         // tensor_dim0[15:0]
  g1[2] = (int)((td0 >> 16) & 0xffffu) | (int)((td1 & 0xffffu) << 16);
  g1[3] = (int)((td1 >> 16) & 0xffffu) | (int)(tile_w << 16);  // tile_dim0
  g1[4] = (int)(tile_h & 0xffffu);              // tile_dim1 (tile_dim2 = 0)
  g1[5] = (int)stride_elems;                    // tensor_dim0_stride[31:0]
  g1[6] = 0;
  g1[7] = 0;
  v4i z4 = {0, 0, 0, 0};                        // groups 2/3 unused (2D tile)
  v8i z8 = {0, 0, 0, 0, 0, 0, 0, 0};
  __builtin_amdgcn_tensor_load_to_lds(g0, g1, z4, z4, z8, 0);
}
#endif

// ---------------- Kernel 1: 1x1 conv (384x128 GEMM) + PReLU + LN -> Q/K/V f16
// One workgroup per (b,t). y[o,f] = sum_c w[o,c]*x[b,c,f,t].
__global__ __launch_bounds__(256, 1)
void k_qkv(const float* __restrict__ x, const float* __restrict__ w,
           const float* __restrict__ bias, const float* __restrict__ alpha_p,
           const float* __restrict__ gamma, const float* __restrict__ beta,
           _Float16* __restrict__ q16, _Float16* __restrict__ k16,
           _Float16* __restrict__ v16) {
  __shared__ _Float16 xT[FREQ_ * LDW_];   // 17 KB, transposed [f][c]
  __shared__ _Float16 wL[96 * LDW_];      // 26 KB, [o-row][c] padded
  __shared__ float    yL[OC3_ * FREQ_];   // 96 KB, [o][f]
  __shared__ float    red[18];
  const int b = blockIdx.x / T_;
  const int t = blockIdx.x % T_;
  const int tid = threadIdx.x, wv = tid >> 5, lane = tid & 31;
  const float alpha = alpha_p[0];

  for (int i = tid; i < CIN_ * FREQ_; i += 256) {
    int f = i >> 7, c = i & 127;   // contiguous LDS writes per row
    xT[f * LDW_ + c] = (_Float16)x[(((long)(b * CIN_ + c)) * FREQ_ + f) * T_ + t];
  }

  for (int oc = 0; oc < 4; ++oc) {          // 4 chunks of 96 output rows
    const int ob = oc * 96;
    for (int i = tid; i < 96 * CIN_; i += 256) {
      int r = i >> 7, c = i & 127;
      wL[r * LDW_ + c] = (_Float16)w[ob * CIN_ + i];
    }
    __syncthreads();

    v8f acc[3] = {};                         // 24 tiles (6x4), 3 per wave
    for (int kc = 0; kc < 4; ++kc) {         // K = 128 in chunks of 32
#pragma unroll
      for (int i = 0; i < 3; ++i) {
        int tile = wv * 3 + i;
        int rt = tile >> 2, ct = tile & 3;
        v16h a  = frag_a (wL + rt * 16 * LDW_ + kc * 32, LDW_);
        v16h bb = frag_bT(xT + ct * 16 * LDW_ + kc * 32, LDW_);
        acc[i] = wmma_f16(a, bb, acc[i]);
      }
    }
    const int cn = lane & 15, mo = (lane >> 4) << 3;
#pragma unroll
    for (int i = 0; i < 3; ++i) {
      int tile = wv * 3 + i;
      int rt = tile >> 2, ct = tile & 3;
#pragma unroll
      for (int r = 0; r < 8; ++r) {
        int row = ob + rt * 16 + r + mo;
        int col = ct * 16 + cn;
        float v = acc[i][r] + bias[row];
        v = (v >= 0.f) ? v : alpha * v;
        yL[row * FREQ_ + col] = v;           // PReLU applied, pre-LN
      }
    }
    __syncthreads();
  }

  // LayerNorm over all 384*64 values of this (b,t)
  float s = 0.f, ss = 0.f;
  for (int i = tid; i < OC3_ * FREQ_; i += 256) {
    float v = yL[i]; s += v; ss += v * v;
  }
  s = wave_sum(s); ss = wave_sum(ss);
  if (lane == 0) { red[wv * 2] = s; red[wv * 2 + 1] = ss; }
  __syncthreads();
  if (tid == 0) {
    float a = 0.f, b2 = 0.f;
    for (int i = 0; i < 8; ++i) { a += red[2 * i]; b2 += red[2 * i + 1]; }
    float mu = a / (float)(OC3_ * FREQ_);
    float var = b2 / (float)(OC3_ * FREQ_) - mu * mu;
    red[16] = mu; red[17] = rsqrtf(var + 1e-5f);
  }
  __syncthreads();
  const float mu = red[16], rs = red[17];

  // Scatter into per-head Q/K/V f16: head h <- channels [48h..48h+48),
  // sel = (o%48)/16 picks q/k/v, d = (o%16)*64 + f.
  for (int i = tid; i < OC3_ * FREQ_; i += 256) {
    int o = i >> 6, f = i & 63;
    float v = (yL[i] - mu) * rs * gamma[i] + beta[i];
    int h = o / 48, r = o % 48;
    int sel = r >> 4, cp = r & 15;
    long dst = ((long)((b * H_ + h) * T_ + t)) * HD_ + cp * 64 + f;
    _Float16 hv = (_Float16)v;
    if (sel == 0)      q16[dst] = hv;
    else if (sel == 1) k16[dst] = hv;
    else               v16[dst] = hv;
  }
}

// ---------------- Kernel 2: causal scores + softmax -> P f16
// One workgroup per (b,h, 64-row q block). Q/K 64x64 tiles DMA'd by the TDM.
__global__ __launch_bounds__(256, 1)
void k_attn(const _Float16* __restrict__ q16, const _Float16* __restrict__ k16,
            _Float16* __restrict__ p16) {
  __shared__ float    sL[64 * 512];   // 128 KB score strip
  __shared__ _Float16 qL[64 * 64];    // 8 KB [q-row][d]
  __shared__ _Float16 kL[64 * 64];    // 8 KB [k-row][d] (== B^T storage)
  const int bh = blockIdx.x >> 3;
  const int qb = blockIdx.x & 7;
  const int tid = threadIdx.x, wv = tid >> 5, lane = tid & 31;
  const _Float16* qbase = q16 + ((long)bh * T_ + qb * 64) * HD_;
  const _Float16* kbase = k16 + (long)bh * T_ * HD_;
#if USE_TDM
  const unsigned qoff = (unsigned)(uintptr_t)(void*)qL;
  const unsigned koff = (unsigned)(uintptr_t)(void*)kL;
#endif

  for (int nt = 0; nt <= qb; ++nt) {         // causal: skip fully-masked tiles
    v8f acc[2] = {};                          // 16 tiles (4x4), 2 per wave
    for (int kc = 0; kc < 16; ++kc) {         // hd = 1024 in chunks of 64
#if USE_TDM
      if (wv == 0) {                          // one DMA pair per workgroup
        tdm_load_2d_f16(qoff, qbase + kc * 64, 64, 64, HD_);
        tdm_load_2d_f16(koff, kbase + (long)(nt * 64) * HD_ + kc * 64, 64, 64, HD_);
        __builtin_amdgcn_s_wait_tensorcnt(0);
      }
#else
      for (int i = tid; i < 64 * 64; i += 256) {
        int rr = i >> 6, dd = i & 63;
        qL[i] = qbase[(long)rr * HD_ + kc * 64 + dd];
        kL[i] = kbase[(long)(nt * 64 + rr) * HD_ + kc * 64 + dd];
      }
#endif
      __syncthreads();
#pragma unroll
      for (int i = 0; i < 2; ++i) {
        int tile = wv * 2 + i;
        int rt = tile >> 2, ct = tile & 3;
#pragma unroll
        for (int kk = 0; kk < 2; ++kk) {
          v16h a  = frag_a (qL + rt * 16 * 64 + kk * 32, 64);
          v16h bb = frag_bT(kL + ct * 16 * 64 + kk * 32, 64); // B = K^T
          acc[i] = wmma_f16(a, bb, acc[i]);
        }
      }
      __syncthreads();
    }
    const int cn = lane & 15, mo = (lane >> 4) << 3;
#pragma unroll
    for (int i = 0; i < 2; ++i) {
      int tile = wv * 2 + i;
      int rt = tile >> 2, ct = tile & 3;
#pragma unroll
      for (int r = 0; r < 8; ++r)
        sL[(rt * 16 + r + mo) * 512 + nt * 64 + ct * 16 + cn] =
            acc[i][r] * 0.03125f;             // hd^-0.5 = 1/32
    }
  }
  __syncthreads();

  // softmax: wave wv owns rows wv*8..wv*8+7; mask col>row by index test.
  for (int rr = wv * 8; rr < wv * 8 + 8; ++rr) {
    const int gq = qb * 64 + rr;
    float m = -3.4e38f;
    for (int c = lane; c < 512; c += 32)
      if (c <= gq) m = fmaxf(m, sL[rr * 512 + c]);
    m = wave_max(m);
    float sum = 0.f;
    for (int c = lane; c < 512; c += 32)
      if (c <= gq) sum += __expf(sL[rr * 512 + c] - m);
    sum = wave_sum(sum);
    const float inv = 1.f / sum;
    _Float16* prow = p16 + ((long)bh * T_ + gq) * (long)T_;
    for (int c = lane; c < 512; c += 32) {
      float p = (c <= gq) ? __expf(sL[rr * 512 + c] - m) * inv : 0.f;
      prow[c] = (_Float16)p;
    }
  }
}

// ---------------- Kernel 3: ctx = P @ V, scattered into proj-input layout
// ctx[b,h,q,d] -> c16[b][16h + d/64][(d%64)*8 + q/64][q%64]
__global__ __launch_bounds__(256, 1)
void k_ctx(const _Float16* __restrict__ p16, const _Float16* __restrict__ v16,
           _Float16* __restrict__ c16) {
  __shared__ _Float16 pL[64 * 64];     // 8 KB [q-row][t']
  __shared__ _Float16 vT[128 * LDV_];  // 18 KB transposed [d][t'] padded
  const int bh = blockIdx.x >> 6;
  const int rem = blockIdx.x & 63;
  const int mt = rem >> 3, ntb = rem & 7;   // 64 q-rows x 128 d-cols tile
  const int b = bh >> 3, h = bh & 7;
  const int tid = threadIdx.x, wv = tid >> 5, lane = tid & 31;
  const _Float16* pbase = p16 + ((long)bh * T_ + mt * 64) * (long)T_;
  const _Float16* vbase = v16 + (long)bh * T_ * HD_ + ntb * 128;

  v8f acc[4] = {};                           // 32 tiles (4x8), 4 per wave
  for (int kc = 0; kc <= mt; ++kc) {         // causal: P is zero beyond kc>mt
    for (int i = tid; i < 64 * 64; i += 256) {
      int rr = i >> 6, cc = i & 63;
      pL[i] = pbase[(long)rr * T_ + kc * 64 + cc];
    }
    for (int i = tid; i < 64 * 128; i += 256) {
      int rr = i >> 7, cc = i & 127;          // coalesced global read in cc
      vT[cc * LDV_ + rr] = vbase[(long)(kc * 64 + rr) * HD_ + cc];
    }
    if (kc < mt)
      __builtin_prefetch(vbase + (long)((kc + 1) * 64 + (tid >> 2)) * HD_, 0, 0);
    __syncthreads();
#pragma unroll
    for (int i = 0; i < 4; ++i) {
      int tile = wv * 4 + i;
      int rt = tile >> 3, ct = tile & 7;
#pragma unroll
      for (int kk = 0; kk < 2; ++kk) {
        v16h a  = frag_a (pL + rt * 16 * 64 + kk * 32, 64);
        v16h bb = frag_bT(vT + ct * 16 * LDV_ + kk * 32, LDV_);
        acc[i] = wmma_f16(a, bb, acc[i]);
      }
    }
    __syncthreads();
  }
  const int cn = lane & 15, mo = (lane >> 4) << 3;
#pragma unroll
  for (int i = 0; i < 4; ++i) {
    int tile = wv * 4 + i;
    int rt = tile >> 3, ct = tile & 7;
#pragma unroll
    for (int r = 0; r < 8; ++r) {
      int gq = mt * 64 + rt * 16 + r + mo;
      int d  = ntb * 128 + ct * 16 + cn;
      int ch = 16 * h + (d >> 6);
      int tt = (d & 63) * 8 + (gq >> 6);
      int ff = gq & 63;
      c16[((long)(b * 128 + ch) * 512 + tt) * 64 + ff] = (_Float16)acc[i][r];
    }
  }
}

// ---------------- Kernel 4: proj 128x128 GEMM + PReLU + LN -> final output
// One workgroup per (b,tt); out[b][o][ff][tt].
__global__ __launch_bounds__(256, 1)
void k_proj(const _Float16* __restrict__ c16, const float* __restrict__ w,
            const float* __restrict__ bias, const float* __restrict__ alpha_p,
            const float* __restrict__ gamma, const float* __restrict__ beta,
            float* __restrict__ out) {
  __shared__ _Float16 wL[128 * LDW_];  // 34 KB padded [o][c]
  __shared__ _Float16 cT[64 * LDW_];   // 17 KB transposed [ff][c]
  __shared__ float    yL[128 * 64];    // 32 KB
  __shared__ float    red[18];
  const int b = blockIdx.x >> 9;
  const int tt = blockIdx.x & 511;
  const int tid = threadIdx.x, wv = tid >> 5, lane = tid & 31;
  const float alpha = alpha_p[0];

  for (int i = tid; i < 128 * 128; i += 256) {
    int r = i >> 7, c = i & 127;
    wL[r * LDW_ + c] = (_Float16)w[i];
  }
  for (int i = tid; i < 128 * 64; i += 256) {
    int c = i >> 6, ff = i & 63;                // coalesced global read in ff
    cT[ff * LDW_ + c] = c16[((long)(b * 128 + c) * 512 + tt) * 64 + ff];
  }
  __syncthreads();

  v8f acc[4] = {};                           // 32 tiles (8x4), 4 per wave
  for (int kc = 0; kc < 4; ++kc) {
#pragma unroll
    for (int i = 0; i < 4; ++i) {
      int tile = wv * 4 + i;
      int rt = tile >> 2, ct = tile & 3;
      v16h a  = frag_a (wL + rt * 16 * LDW_ + kc * 32, LDW_);
      v16h bb = frag_bT(cT + ct * 16 * LDW_ + kc * 32, LDW_);
      acc[i] = wmma_f16(a, bb, acc[i]);
    }
  }
  const int cn = lane & 15, mo = (lane >> 4) << 3;
#pragma unroll
  for (int i = 0; i < 4; ++i) {
    int tile = wv * 4 + i;
    int rt = tile >> 2, ct = tile & 3;
#pragma unroll
    for (int r = 0; r < 8; ++r) {
      int row = rt * 16 + r + mo, col = ct * 16 + cn;
      float v = acc[i][r] + bias[row];
      v = (v >= 0.f) ? v : alpha * v;
      yL[row * 64 + col] = v;
    }
  }
  __syncthreads();

  float s = 0.f, ss = 0.f;
  for (int i = tid; i < 128 * 64; i += 256) {
    float v = yL[i]; s += v; ss += v * v;
  }
  s = wave_sum(s); ss = wave_sum(ss);
  if (lane == 0) { red[wv * 2] = s; red[wv * 2 + 1] = ss; }
  __syncthreads();
  if (tid == 0) {
    float a = 0.f, b2 = 0.f;
    for (int i = 0; i < 8; ++i) { a += red[2 * i]; b2 += red[2 * i + 1]; }
    float mu = a / 8192.f;
    float var = b2 / 8192.f - mu * mu;
    red[16] = mu; red[17] = rsqrtf(var + 1e-5f);
  }
  __syncthreads();
  const float mu = red[16], rs = red[17];
  for (int i = tid; i < 128 * 64; i += 256) {
    int o = i >> 6, ff = i & 63;
    float v = (yL[i] - mu) * rs * gamma[i] + beta[i];
    out[((long)(b * 128 + o) * 64 + ff) * 512 + tt] = v;   // final swapaxes(2,3)
  }
}

// ---------------- Host launcher
extern "C" void kernel_launch(void* const* d_in, const int* in_sizes, int n_in,
                              void* d_out, int out_size, void* d_ws, size_t ws_size,
                              hipStream_t stream) {
  const float* x          = (const float*)d_in[0];
  const float* qkv_w      = (const float*)d_in[1];
  const float* qkv_b      = (const float*)d_in[2];
  const float* qkv_alpha  = (const float*)d_in[3];
  const float* qkv_gamma  = (const float*)d_in[4];
  const float* qkv_beta   = (const float*)d_in[5];
  const float* proj_w     = (const float*)d_in[6];
  const float* proj_b     = (const float*)d_in[7];
  const float* proj_alpha = (const float*)d_in[8];
  const float* proj_gamma = (const float*)d_in[9];
  const float* proj_beta  = (const float*)d_in[10];
  float* out = (float*)d_out;

  // Workspace layout (f16): Q | K | V | P ; ctx (c16) aliases Q (dead after k_attn).
  const size_t QS = (size_t)B_ * H_ * T_ * HD_ * sizeof(_Float16);  // 33.5 MB
  char* ws = (char*)d_ws;
  _Float16* q16 = (_Float16*)(ws);
  _Float16* k16 = (_Float16*)(ws + QS);
  _Float16* v16 = (_Float16*)(ws + 2 * QS);
  _Float16* p16 = (_Float16*)(ws + 3 * QS);
  _Float16* c16 = q16;  // alias: Q no longer read once P exists

  k_qkv<<<dim3(B_ * T_), dim3(256), 0, stream>>>(
      x, qkv_w, qkv_b, qkv_alpha, qkv_gamma, qkv_beta, q16, k16, v16);
  k_attn<<<dim3(B_ * H_ * (T_ / 64)), dim3(256), 0, stream>>>(q16, k16, p16);
  k_ctx<<<dim3(B_ * H_ * (T_ / 64) * (HD_ / 128)), dim3(256), 0, stream>>>(
      p16, v16, c16);
  k_proj<<<dim3(B_ * T_), dim3(256), 0, stream>>>(
      c16, proj_w, proj_b, proj_alpha, proj_gamma, proj_beta, out);
}
